// MultiHeadAttention_31825707664152
// MI455X (gfx1250) — matrix operand order repaired and checked by reading, not compiled, not run on hardware
//
#include <hip/hip_runtime.h>
#include <hip/hip_bf16.h>

// ---------------------------------------------------------------------------
// Complex multi-head attention, CDNA5 (gfx1250) wave32 + WMMA bf16 pipeline.
//   B=2, NQ=NK=2048, R=512, H=8, D=64
// Stage 0: one-shot f32->bf16 packing of activations and complex weights
// Stage 1: pure-bf16 complex projections (WMMA, software-pipelined loads)
// Stage 2: fused flash-style complex-score attention -> packed [o_re|o_im]
// Stage 3: pure-bf16 complex output projection -> f32 (B,NQ,R,2)
// ---------------------------------------------------------------------------

typedef __attribute__((ext_vector_type(16))) __bf16        v16bf;
typedef __attribute__((ext_vector_type(8)))  __bf16        v8bf;
typedef __attribute__((ext_vector_type(4)))  __bf16        v4bf;
typedef __attribute__((ext_vector_type(8)))  float         v8f;
typedef __attribute__((ext_vector_type(8)))  unsigned int  v8u;

constexpr int kB  = 2;
constexpr int kNQ = 2048;
constexpr int kNK = 2048;
constexpr int kR  = 512;
constexpr int kH  = 8;
constexpr int kD  = 64;
constexpr int kM  = kB * kNQ;      // 4096 rows for the projection GEMMs
constexpr int kKp = 2 * kR;        // 1024 packed K (re|im concat)

// ---------------- WMMA helpers ----------------

static __device__ __forceinline__ v8f wmma_bf16(v16bf a, v16bf b, v8f c) {
  // D = A(16x32 bf16) * B(32x16 bf16) + C(16x16 f32)
  return __builtin_amdgcn_wmma_f32_16x16x32_bf16(false, a, false, b,
                                                 (short)0, c, false, false);
}

static __device__ __forceinline__ v16bf neg_bf16x16(v16bf x) {
  v8u u = __builtin_bit_cast(v8u, x);
  u ^= 0x80008000u;
  return __builtin_bit_cast(v16bf, u);
}

// A-matrix fragment (16x32, bf16, row-major source, stride lda elements).
// ISA layout: lane L holds row (L&15); half-wave selects K 0-7 vs 8-15 in
// VGPRs 0-3 and K 16-23 vs 24-31 in VGPRs 4-7.
static __device__ __forceinline__ v16bf load_a_bf16(const __bf16* a0, int lda,
                                                    int lane) {
  const int row = lane & 15;
  const int kb  = (lane >> 4) << 3;   // 0 or 8
  const __bf16* p = a0 + (size_t)row * lda;
  v8bf lo = *(const v8bf*)(p + kb);
  v8bf hi = *(const v8bf*)(p + 16 + kb);
  v16bf r;
#pragma unroll
  for (int j = 0; j < 8; ++j) { r[j] = lo[j]; r[8 + j] = hi[j]; }
  return r;
}

// B-matrix fragment (32x16) loaded from the TRANSPOSED operand Bt (16xK,
// row-major, stride ldbt): lane L holds column (L&15); half-wave selects
// K 0-7 vs 8-15 in VGPRs 0-3 and K 16-23 vs 24-31 in VGPRs 4-7.
static __device__ __forceinline__ v16bf load_bt_bf16(const __bf16* bt, int ldbt,
                                                     int lane) {
  const int n  = lane & 15;
  const int kb = (lane >> 4) << 3;    // 0 or 8
  const __bf16* p = bt + (size_t)n * ldbt + kb;
  v8bf lo = *(const v8bf*)(p);
  v8bf hi = *(const v8bf*)(p + 16);
  v16bf r;
#pragma unroll
  for (int j = 0; j < 8; ++j) { r[j] = lo[j]; r[8 + j] = hi[j]; }
  return r;
}

// C/D layout: lane L holds column (L&15); VGPR r holds row r + 8*(L>>4).
static __device__ __forceinline__ void store_c_bf16(__bf16* d, int ldd,
                                                    int lane, v8f c) {
  const int col = lane & 15;
  const int rb  = (lane >> 4) << 3;
#pragma unroll
  for (int r = 0; r < 8; ++r) d[(size_t)(rb + r) * ldd + col] = (__bf16)c[r];
}

// ---------------- Stage 0a: pack activations [xr | xi] -> bf16 --------------
// xc: (4096 x 1024); each thread converts 4 consecutive f32 of one half.
__global__ __launch_bounds__(256) void cvt_pack_kernel(
    const float* __restrict__ xr, const float* __restrict__ xi,
    __bf16* __restrict__ xc) {
  const long t    = (long)blockIdx.x * 256 + threadIdx.x;
  const long half = (long)kM * kR / 4;           // 524288 quads per half
  const int  isIm = (t >= half);
  const float* src = isIm ? xi : xr;
  const long e = (isIm ? (t - half) : t) * 4;    // element in (4096 x 512)
  const long row = e >> 9;                       // /512
  const long col = e & 511;
  const float4 v = *(const float4*)(src + e);
  v4bf o;
  o[0] = (__bf16)v.x; o[1] = (__bf16)v.y; o[2] = (__bf16)v.z; o[3] = (__bf16)v.w;
  *(v4bf*)(xc + row * kKp + (isIm ? kR : 0) + col) = o;
}

// ---------------- Stage 0b: pack complex weights -> bf16 --------------------
// w0 = [wr | -wi], w1 = [wi | wr]; each (512 x 1024) from (512 x 512) halves.
__global__ __launch_bounds__(256) void pack_w_kernel(
    const float* __restrict__ wr, const float* __restrict__ wi,
    __bf16* __restrict__ w0, __bf16* __restrict__ w1) {
  const long t = (long)blockIdx.x * 256 + threadIdx.x;
  const long e = t * 4;                          // element in (512 x 512)
  const long row = e >> 9;
  const long col = e & 511;
  const float4 r = *(const float4*)(wr + e);
  const float4 i = *(const float4*)(wi + e);
  v4bf br, bi, bni;
  br[0]  = (__bf16)r.x;  br[1]  = (__bf16)r.y;  br[2]  = (__bf16)r.z;  br[3]  = (__bf16)r.w;
  bi[0]  = (__bf16)i.x;  bi[1]  = (__bf16)i.y;  bi[2]  = (__bf16)i.z;  bi[3]  = (__bf16)i.w;
  bni[0] = (__bf16)-i.x; bni[1] = (__bf16)-i.y; bni[2] = (__bf16)-i.z; bni[3] = (__bf16)-i.w;
  *(v4bf*)(w0 + row * kKp + col)      = br;    // [wr | .]
  *(v4bf*)(w0 + row * kKp + kR + col) = bni;   // [. | -wi]
  *(v4bf*)(w1 + row * kKp + col)      = bi;    // [wi | .]
  *(v4bf*)(w1 + row * kKp + kR + col) = br;    // [. | wr]
}

// ---------------- complex GEMM core (pipelined fragment sets) ---------------

struct Frags {
  v16bf a;
  v16bf b0[4];
  v16bf b1[4];
};

static __device__ __forceinline__ void load_frags(
    Frags& f, const __bf16* __restrict__ Xc, const __bf16* __restrict__ W0,
    const __bf16* __restrict__ W1, int M, int Nw, int c, int lane) {
  f.a = load_a_bf16(Xc + (size_t)M * kKp + c * 32, kKp, lane);
#pragma unroll
  for (int t = 0; t < 4; ++t) {
    const size_t wb = (size_t)(Nw + t * 16) * kKp + c * 32;
    f.b0[t] = load_bt_bf16(W0 + wb, kKp, lane);
    f.b1[t] = load_bt_bf16(W1 + wb, kKp, lane);
  }
}

static __device__ __forceinline__ void mul_frags(const Frags& f, v8f (&ar)[4],
                                                 v8f (&ai)[4]) {
#pragma unroll
  for (int t = 0; t < 4; ++t) {
    ar[t] = wmma_bf16(f.a, f.b0[t], ar[t]);
    ai[t] = wmma_bf16(f.a, f.b1[t], ai[t]);
  }
}

// ---------------- Stage 1: complex projection (pure bf16 WMMA) --------------
// Yre = Xc @ W0^T, Yim = Xc @ W1^T. Wave tile: 16M x 64N per part; fragment
// loads for chunk c+1 are issued before the WMMAs of chunk c (ping-pong).
__global__ __launch_bounds__(128) void cgemm_proj_kernel(
    const __bf16* __restrict__ Xc,   // (4096 x 1024)
    const __bf16* __restrict__ W0,   // (512 x 1024)
    const __bf16* __restrict__ W1,   // (512 x 1024)
    __bf16* __restrict__ Yre, __bf16* __restrict__ Yim) {
  const int lane = threadIdx.x & 31;
  const int wv   = threadIdx.x >> 5;
  const int M    = blockIdx.x * 16;
  const int Nw   = (blockIdx.y * 4 + wv) * 64;
  v8f ar[4] = {};
  v8f ai[4] = {};
  Frags fA, fB;
  load_frags(fA, Xc, W0, W1, M, Nw, 0, lane);
#pragma unroll 4
  for (int c = 0; c < kKp / 32; c += 2) {          // 32 chunks, 2 per iter
    load_frags(fB, Xc, W0, W1, M, Nw, c + 1, lane);
    mul_frags(fA, ar, ai);                         // overlaps fB loads
    if (c + 2 < kKp / 32)
      load_frags(fA, Xc, W0, W1, M, Nw, c + 2, lane);
    mul_frags(fB, ar, ai);                         // overlaps fA loads
  }
#pragma unroll
  for (int t = 0; t < 4; ++t) {
    store_c_bf16(Yre + (size_t)M * kR + Nw + t * 16, kR, lane, ar[t]);
    store_c_bf16(Yim + (size_t)M * kR + Nw + t * 16, kR, lane, ai[t]);
  }
}

// ---------------- Stage 2: fused complex attention ----------------
// 8 waves / WG, each wave owns 16 queries of one (b,h); streams 32-key tiles.
__global__ __launch_bounds__(256) void attn_kernel(
    const __bf16* __restrict__ qre, const __bf16* __restrict__ qim,
    const __bf16* __restrict__ kre, const __bf16* __restrict__ kim,
    const __bf16* __restrict__ vre, const __bf16* __restrict__ vim,
    __bf16* __restrict__ Oc) {
  constexpr int ldk = kD + 8;   // 72, padded K-tile row (32 x 64 bf16)
  constexpr int ldv = 32 + 8;   // 40, padded V^T row   (64 x 32 bf16)
  constexpr int ldp = 32 + 8;   // 40, padded P row     (16 x 32 bf16)
  __shared__ __bf16 sKr[32 * ldk];
  __shared__ __bf16 sKi[32 * ldk];
  __shared__ __bf16 sVr[kD * ldv];
  __shared__ __bf16 sVi[kD * ldv];
  __shared__ __bf16 sP[8][16 * ldp];

  const int tid  = threadIdx.x;
  const int lane = tid & 31;
  const int wv   = tid >> 5;
  const int h    = blockIdx.y;
  const int b    = blockIdx.z;
  const int qrow0 = b * kNQ + blockIdx.x * 128 + wv * 16;
  const int hcol  = h * kD;

  // Q fragments (d split 0-31 / 32-63), held for the whole key stream.
  const size_t qoff = (size_t)qrow0 * kR + hcol;
  v16bf qr0 = load_a_bf16(qre + qoff,      kR, lane);
  v16bf qr1 = load_a_bf16(qre + qoff + 32, kR, lane);
  v16bf qi0 = load_a_bf16(qim + qoff,      kR, lane);
  v16bf qi1 = load_a_bf16(qim + qoff + 32, kR, lane);

  v8f o_re[4] = {};
  v8f o_im[4] = {};
  float mrow[8], lrow[8];
#pragma unroll
  for (int r = 0; r < 8; ++r) { mrow[r] = -3.0e38f; lrow[r] = 0.f; }

  const float scale = 0.125f;   // 1/sqrt(64)
  const int   ntile = kNK / 32; // 64

  for (int kt = 0; kt < ntile; ++kt) {
    // -------- cooperative K/V tile load (32 keys x 64 d), V transposed -----
    {
      const int row = tid >> 3;          // 0..31
      const int cg  = (tid & 7) * 8;     // 0..56
      const size_t g = ((size_t)(b * kNK + kt * 32 + row)) * kR + hcol + cg;
      *(v8bf*)&sKr[row * ldk + cg] = *(const v8bf*)(kre + g);
      *(v8bf*)&sKi[row * ldk + cg] = *(const v8bf*)(kim + g);
      v8bf vr8 = *(const v8bf*)(vre + g);
      v8bf vi8 = *(const v8bf*)(vim + g);
#pragma unroll
      for (int j = 0; j < 8; ++j) {
        sVr[(cg + j) * ldv + row] = vr8[j];
        sVi[(cg + j) * ldv + row] = vi8[j];
      }
      if (kt + 1 < ntile) {
        const size_t gn = g + (size_t)32 * kR;
        __builtin_prefetch(kre + gn, 0, 1);
        __builtin_prefetch(kim + gn, 0, 1);
        __builtin_prefetch(vre + gn, 0, 1);
        __builtin_prefetch(vim + gn, 0, 1);
      }
    }
    __syncthreads();

    // -------- complex scores: preload all 8 K fragments, then 16 WMMAs -----
    v16bf kf[2][4];   // [subtile j][kr0, kr1, ki0, ki1]
#pragma unroll
    for (int j = 0; j < 2; ++j) {
      const __bf16* kbR = &sKr[j * 16 * ldk];
      const __bf16* kbI = &sKi[j * 16 * ldk];
      kf[j][0] = load_bt_bf16(kbR,      ldk, lane);
      kf[j][1] = load_bt_bf16(kbR + 32, ldk, lane);
      kf[j][2] = load_bt_bf16(kbI,      ldk, lane);
      kf[j][3] = load_bt_bf16(kbI + 32, ldk, lane);
    }
    float mag2[2][8];
#pragma unroll
    for (int j = 0; j < 2; ++j) {
      v8f sre = {};
      sre = wmma_bf16(qr0, kf[j][0], sre);
      sre = wmma_bf16(qr1, kf[j][1], sre);
      sre = wmma_bf16(qi0, kf[j][2], sre);
      sre = wmma_bf16(qi1, kf[j][3], sre);

      v8f sim = {};
      sim = wmma_bf16(qi0, kf[j][0], sim);
      sim = wmma_bf16(qi1, kf[j][1], sim);
      sim = wmma_bf16(qr0, neg_bf16x16(kf[j][2]), sim);
      sim = wmma_bf16(qr1, neg_bf16x16(kf[j][3]), sim);

#pragma unroll
      for (int r = 0; r < 8; ++r)
        mag2[j][r] = sqrtf(sre[r] * sre[r] + sim[r] * sim[r]) * scale;
    }

    // -------- online softmax over the 32-key tile -------------------------
    const int col = lane & 15;
    const int rb  = (lane >> 4) << 3;
#pragma unroll
    for (int r = 0; r < 8; ++r) {
      float tm = fmaxf(mag2[0][r], mag2[1][r]);
#pragma unroll
      for (int off = 1; off < 16; off <<= 1)
        tm = fmaxf(tm, __shfl_xor(tm, off, 32));   // half-wave row reduce
      const float mn = fmaxf(mrow[r], tm);
      const float alpha = __expf(mrow[r] - mn);
      mrow[r] = mn;
      const float p0 = __expf(mag2[0][r] - mn);
      const float p1 = __expf(mag2[1][r] - mn);
      float rs = p0 + p1;
#pragma unroll
      for (int off = 1; off < 16; off <<= 1)
        rs += __shfl_xor(rs, off, 32);
      lrow[r] = lrow[r] * alpha + rs;
#pragma unroll
      for (int t = 0; t < 4; ++t) { o_re[t][r] *= alpha; o_im[t][r] *= alpha; }
      sP[wv][(rb + r) * ldp + col]      = (__bf16)p0;   // C-layout -> LDS
      sP[wv][(rb + r) * ldp + 16 + col] = (__bf16)p1;
    }

    // -------- P @ V: preload P + all 8 V fragments, then 8 WMMAs -----------
    v16bf ap = load_a_bf16(&sP[wv][0], ldp, lane);
    v16bf bvr[4], bvi[4];
#pragma unroll
    for (int t = 0; t < 4; ++t) {
      bvr[t] = load_bt_bf16(&sVr[t * 16 * ldv], ldv, lane);
      bvi[t] = load_bt_bf16(&sVi[t * 16 * ldv], ldv, lane);
    }
#pragma unroll
    for (int t = 0; t < 4; ++t) {
      o_re[t] = wmma_bf16(ap, bvr[t], o_re[t]);
      o_im[t] = wmma_bf16(ap, bvi[t], o_im[t]);
    }
    __syncthreads();
  }

  // -------- epilogue: normalize, pack [o_re | o_im] bf16 --------------------
  float invl[8];
#pragma unroll
  for (int r = 0; r < 8; ++r) invl[r] = 1.0f / lrow[r];
  const int col = lane & 15;
  const int rb  = (lane >> 4) << 3;
#pragma unroll
  for (int t = 0; t < 4; ++t)
#pragma unroll
    for (int r = 0; r < 8; ++r) {
      const size_t row = (size_t)(qrow0 + rb + r);
      const int d = t * 16 + col;
      Oc[row * kKp + hcol + d]      = (__bf16)(o_re[t][r] * invl[r]);
      Oc[row * kKp + kR + hcol + d] = (__bf16)(o_im[t][r] * invl[r]);
    }
}

// ---------------- Stage 3: complex output projection (pure bf16 WMMA) -------
// out_re = Oc @ W0^T, out_im = Oc @ W1^T, interleaved f32 (.., R, 2).
__global__ __launch_bounds__(128) void cgemm_out_kernel(
    const __bf16* __restrict__ Oc,   // (4096 x 1024) = [o_re | o_im]
    const __bf16* __restrict__ W0, const __bf16* __restrict__ W1,
    float* __restrict__ out) {
  const int lane = threadIdx.x & 31;
  const int wv   = threadIdx.x >> 5;
  const int M    = blockIdx.x * 16;
  const int Nw   = (blockIdx.y * 4 + wv) * 64;
  v8f ar[4] = {};
  v8f ai[4] = {};
  Frags fA, fB;
  load_frags(fA, Oc, W0, W1, M, Nw, 0, lane);
#pragma unroll 4
  for (int c = 0; c < kKp / 32; c += 2) {
    load_frags(fB, Oc, W0, W1, M, Nw, c + 1, lane);
    mul_frags(fA, ar, ai);
    if (c + 2 < kKp / 32)
      load_frags(fA, Oc, W0, W1, M, Nw, c + 2, lane);
    mul_frags(fB, ar, ai);
  }
  const int col = lane & 15;
  const int rb  = (lane >> 4) << 3;
#pragma unroll
  for (int t = 0; t < 4; ++t)
#pragma unroll
    for (int r = 0; r < 8; ++r) {
      const size_t o = ((size_t)(M + rb + r)) * kR + (Nw + t * 16 + col);
      out[o * 2 + 0] = ar[t][r];
      out[o * 2 + 1] = ai[t][r];
    }
}

// ---------------- host-side orchestration ----------------
extern "C" void kernel_launch(void* const* d_in, const int* in_sizes, int n_in,
                              void* d_out, int out_size, void* d_ws,
                              size_t ws_size, hipStream_t stream) {
  (void)in_sizes; (void)n_in; (void)out_size; (void)ws_size;
  const float* Qr  = (const float*)d_in[0];
  const float* Qi  = (const float*)d_in[1];
  const float* Kr  = (const float*)d_in[2];
  const float* Ki  = (const float*)d_in[3];
  const float* Vr  = (const float*)d_in[4];
  const float* Vi  = (const float*)d_in[5];
  const float* wqr = (const float*)d_in[6];
  const float* wqi = (const float*)d_in[7];
  const float* wkr = (const float*)d_in[8];
  const float* wki = (const float*)d_in[9];
  const float* wvr = (const float*)d_in[10];
  const float* wvi = (const float*)d_in[11];
  const float* wor = (const float*)d_in[12];
  const float* woi = (const float*)d_in[13];

  char* ws = (char*)d_ws;
  const size_t X  = (size_t)kM * kKp * sizeof(__bf16);   // 8 MiB packed acts
  const size_t WU = (size_t)kR * kKp * sizeof(__bf16);   // 1 MiB packed weight
  const size_t P  = (size_t)kM * kR  * sizeof(__bf16);   // 4 MiB projection
  __bf16* Xq  = (__bf16*)(ws + 0 * X);
  __bf16* Xk  = (__bf16*)(ws + 1 * X);
  __bf16* Xv  = (__bf16*)(ws + 2 * X);
  char*   wb  = ws + 3 * X;
  __bf16* Wq0 = (__bf16*)(wb + 0 * WU);
  __bf16* Wq1 = (__bf16*)(wb + 1 * WU);
  __bf16* Wk0 = (__bf16*)(wb + 2 * WU);
  __bf16* Wk1 = (__bf16*)(wb + 3 * WU);
  __bf16* Wv0 = (__bf16*)(wb + 4 * WU);
  __bf16* Wv1 = (__bf16*)(wb + 5 * WU);
  __bf16* Wo0 = (__bf16*)(wb + 6 * WU);
  __bf16* Wo1 = (__bf16*)(wb + 7 * WU);
  char*   pb  = wb + 8 * WU;
  __bf16* q_re = (__bf16*)(pb + 0 * P);
  __bf16* q_im = (__bf16*)(pb + 1 * P);
  __bf16* k_re = (__bf16*)(pb + 2 * P);
  __bf16* k_im = (__bf16*)(pb + 3 * P);
  __bf16* v_re = (__bf16*)(pb + 4 * P);
  __bf16* v_im = (__bf16*)(pb + 5 * P);
  __bf16* Oc   = (__bf16*)(pb + 6 * P);                  // 8 MiB packed O

  // Stage 0: pack/convert (bandwidth-bound one-shots)
  const int cvtBlocks = 2 * (kM * kR / 4) / 256;         // 4096
  cvt_pack_kernel<<<cvtBlocks, 256, 0, stream>>>(Qr, Qi, Xq);
  cvt_pack_kernel<<<cvtBlocks, 256, 0, stream>>>(Kr, Ki, Xk);
  cvt_pack_kernel<<<cvtBlocks, 256, 0, stream>>>(Vr, Vi, Xv);
  const int wBlocks = (kR * kR / 4) / 256;               // 256
  pack_w_kernel<<<wBlocks, 256, 0, stream>>>(wqr, wqi, Wq0, Wq1);
  pack_w_kernel<<<wBlocks, 256, 0, stream>>>(wkr, wki, Wk0, Wk1);
  pack_w_kernel<<<wBlocks, 256, 0, stream>>>(wvr, wvi, Wv0, Wv1);
  pack_w_kernel<<<wBlocks, 256, 0, stream>>>(wor, woi, Wo0, Wo1);

  // Stage 1: complex projections
  const dim3 gg(kM / 16, kR / 256, 1);   // (256, 2)
  const dim3 gb(128, 1, 1);
  cgemm_proj_kernel<<<gg, gb, 0, stream>>>(Xq, Wq0, Wq1, q_re, q_im);
  cgemm_proj_kernel<<<gg, gb, 0, stream>>>(Xk, Wk0, Wk1, k_re, k_im);
  cgemm_proj_kernel<<<gg, gb, 0, stream>>>(Xv, Wv0, Wv1, v_re, v_im);

  // Stage 2: fused attention
  attn_kernel<<<dim3(kNQ / 128, kH, kB), dim3(256, 1, 1), 0, stream>>>(
      q_re, q_im, k_re, k_im, v_re, v_im, Oc);

  // Stage 3: output projection
  cgemm_out_kernel<<<gg, gb, 0, stream>>>(Oc, Wo0, Wo1, (float*)d_out);
}
